// EGNNLayer_73993696575521
// MI455X (gfx1250) — matrix operand order repaired and checked by reading, not compiled, not run on hardware
//
#include <hip/hip_runtime.h>

#define H 128
#define EPS 1e-8f
#define LN_EPS 1e-5f

typedef __attribute__((ext_vector_type(16))) __bf16 v16bf;
typedef __attribute__((ext_vector_type(8)))  __bf16 v8bf;
typedef __attribute__((ext_vector_type(4)))  __bf16 v4bf;
typedef __attribute__((ext_vector_type(8)))  float  v8f;

__device__ __forceinline__ void lds_fence() {
  // DS ops are issued in-order per wave; wait for LDS writes to land before
  // cross-lane reads, and stop the compiler from reordering around it.
  asm volatile("s_wait_dscnt 0" ::: "memory");
}

__device__ __forceinline__ float silu_f(float x) {
  return x / (1.0f + __expf(-x));
}

// C-frag layout (16x16 f32, wave32): lane l, vgpr v -> M = v + 8*(l>>4), N = l&15.
// A-frag layout (16x32 bf16): lane l -> M = l&15; K runs [8*hi,8*hi+8) and
// [16+8*hi, 16+8*hi+8) with hi = l>>4.  B frags are pre-packed in global memory
// so lane l reads its 16 contiguous bf16 values.
__device__ __forceinline__ v8f wmma_tiles(const __bf16* As, int Kdim,
                                          const __bf16* Bp, int ntile,
                                          int nks, v8f c, int lane) {
  const int hi = lane >> 4;
  const int M  = lane & 15;
#pragma unroll
  for (int ks = 0; ks < nks; ++ks) {
    v8bf a0 = *(const v8bf*)(As + M * Kdim + ks * 32 + 8 * hi);
    v8bf a1 = *(const v8bf*)(As + M * Kdim + ks * 32 + 16 + 8 * hi);
    v16bf a;
#pragma unroll
    for (int i = 0; i < 8; ++i) { a[i] = a0[i]; a[8 + i] = a1[i]; }
    const __bf16* bl = Bp + (size_t)(((ks << 3) + ntile) * 32 + lane) * 16;
    v8bf b0 = *(const v8bf*)bl;
    v8bf b1 = *(const v8bf*)(bl + 8);
    v16bf b;
#pragma unroll
    for (int i = 0; i < 8; ++i) { b[i] = b0[i]; b[8 + i] = b1[i]; }
    c = __builtin_amdgcn_wmma_f32_16x16x32_bf16(false, a, false, b, (short)0, c,
                                                false, false);
  }
  return c;
}

// Re-pack a row-major fp32 weight [Kdim x 128] into bf16 B-fragment order:
// frag(ks,nt) -> 32 lanes x 16 contiguous values; lane l holds
// K = ks*32 + 8*(l>>4) + (j&7) + 16*(j>>3) (j=0..15), N = nt*16 + (l&15).
__global__ void egnn_pack_w(const float* __restrict__ W, __bf16* __restrict__ out,
                            int Kdim) {
  int idx = blockIdx.x * 256 + threadIdx.x;
  if (idx >= Kdim * H) return;
  int frag = idx >> 9;
  int rem  = idx & 511;
  int lane = rem >> 4;
  int j    = rem & 15;
  int ks = frag >> 3;
  int nt = frag & 7;
  int K = ks * 32 + 8 * (lane >> 4) + (j & 7) + 16 * (j >> 3);
  int N = nt * 16 + (lane & 15);
  out[idx] = (__bf16)W[(size_t)K * H + N];
}

__global__ __launch_bounds__(128) void egnn_edge_kernel(
    const float* __restrict__ h, const float* __restrict__ pos,
    const long long* __restrict__ ei,
    const float* __restrict__ ew1, const float* __restrict__ eb1,
    const float* __restrict__ eb2, const float* __restrict__ cb1,
    const float* __restrict__ cw2,
    const __bf16* __restrict__ ew1p, const __bf16* __restrict__ ew2p,
    const __bf16* __restrict__ cw1p,
    float* __restrict__ aggMsg, float* __restrict__ aggPos,
    int E, int nTiles) {
  __shared__ __bf16 Ast_s[4][16 * 256];   // edge-feature stage, later m_ij stage
  __shared__ __bf16 Ms_s[4][16 * 128];    // layer-1 output stage
  __shared__ float  cw_s[4][16];

  const int lane = threadIdx.x & 31;
  const int wid  = threadIdx.x >> 5;
  const int tile = blockIdx.x * 4 + wid;
  if (tile >= nTiles) return;

  __bf16* Ast = &Ast_s[wid][0];
  __bf16* Ms  = &Ms_s[wid][0];
  const int hi = lane >> 4;
  const int lo = lane & 15;

  int rowI = 0, colI = 0;
  float dx = 0.f, dy = 0.f, dz = 0.f, dist = 0.f;
  if (lane < 16) {
    int e = tile * 16 + lane;
    if (e >= E) e = E - 1;
    rowI = (int)ei[e];
    colI = (int)ei[(size_t)E + e];
    dx = pos[rowI * 3 + 0] - pos[colI * 3 + 0];
    dy = pos[rowI * 3 + 1] - pos[colI * 3 + 1];
    dz = pos[rowI * 3 + 2] - pos[colI * 3 + 2];
    dist = sqrtf(dx * dx + dy * dy + dz * dz + EPS);
  }

  // Gather [h[row] | h[col]] (fp32, L2-resident) -> bf16 LDS stage [16][256]
#pragma unroll 4
  for (int it = 0; it < 32; ++it) {
    int c4   = it * 32 + lane;    // 1024 float4 chunks per tile
    int eIdx = c4 >> 6;
    int kc   = c4 & 63;
    int r  = __shfl(rowI, eIdx);
    int cI = __shfl(colI, eIdx);
    const float4* src = (kc < 32)
        ? (const float4*)(h + (size_t)r * H + kc * 4)
        : (const float4*)(h + (size_t)cI * H + (kc - 32) * 4);
    float4 v = *src;
    v4bf o = { (__bf16)v.x, (__bf16)v.y, (__bf16)v.z, (__bf16)v.w };
    *(v4bf*)(Ast + eIdx * 256 + kc * 4) = o;
  }
  lds_fence();

  // edge layer 1: silu(feat @ ew1 + eb1); dist column folded into C init (fp32)
#pragma unroll
  for (int nt = 0; nt < 8; ++nt) {
    int Ncol = nt * 16 + lo;
    float bval = eb1[Ncol];
    float w256 = ew1[256 * H + Ncol];
    v8f c;
#pragma unroll
    for (int v = 0; v < 8; ++v)
      c[v] = bval + __shfl(dist, v + 8 * hi) * w256;
    c = wmma_tiles(Ast, 256, ew1p, nt, 8, c, lane);
#pragma unroll
    for (int v = 0; v < 8; ++v)
      Ms[(v + 8 * hi) * H + Ncol] = (__bf16)silu_f(c[v]);
  }
  lds_fence();

  // edge layer 2 -> m_ij; fp32 scatter-add into aggMsg; bf16 restage into Ast
#pragma unroll
  for (int nt = 0; nt < 8; ++nt) {
    int Ncol = nt * 16 + lo;
    float b2 = eb2[Ncol];
    v8f c;
#pragma unroll
    for (int v = 0; v < 8; ++v) c[v] = b2;
    c = wmma_tiles(Ms, 128, ew2p, nt, 4, c, lane);
#pragma unroll
    for (int v = 0; v < 8; ++v) {
      float s  = silu_f(c[v]);
      int  Mr  = v + 8 * hi;
      int  rM  = __shfl(rowI, Mr);
      if (tile * 16 + Mr < E)
        unsafeAtomicAdd(&aggMsg[(size_t)rM * H + Ncol], s);
      Ast[Mr * H + Ncol] = (__bf16)s;
    }
  }
  lds_fence();

  // coord gate: (silu(m_ij @ cw1 + cb1)) @ cw2  -> one scalar per edge
  float pv[8];
#pragma unroll
  for (int v = 0; v < 8; ++v) pv[v] = 0.f;
#pragma unroll
  for (int nt = 0; nt < 8; ++nt) {
    int Ncol = nt * 16 + lo;
    float b1 = cb1[Ncol];
    v8f c;
#pragma unroll
    for (int v = 0; v < 8; ++v) c[v] = b1;
    c = wmma_tiles(Ast, 128, cw1p, nt, 4, c, lane);
    float w2 = cw2[Ncol];
#pragma unroll
    for (int v = 0; v < 8; ++v) pv[v] += silu_f(c[v]) * w2;
  }
#pragma unroll
  for (int v = 0; v < 8; ++v) {
    pv[v] += __shfl_xor(pv[v], 1);
    pv[v] += __shfl_xor(pv[v], 2);
    pv[v] += __shfl_xor(pv[v], 4);
    pv[v] += __shfl_xor(pv[v], 8);
  }
  if (lo == 0) {
#pragma unroll
    for (int v = 0; v < 8; ++v) cw_s[wid][v + 8 * hi] = pv[v];
  }
  lds_fence();

  if (lane < 16 && tile * 16 + lane < E) {
    float cwv = cw_s[wid][lane];
    float inv = 1.f / (dist + EPS);
    unsafeAtomicAdd(&aggPos[(size_t)rowI * 3 + 0], cwv * dx * inv);
    unsafeAtomicAdd(&aggPos[(size_t)rowI * 3 + 1], cwv * dy * inv);
    unsafeAtomicAdd(&aggPos[(size_t)rowI * 3 + 2], cwv * dz * inv);
  }
}

__global__ __launch_bounds__(128) void egnn_node_kernel(
    const float* __restrict__ h, const float* __restrict__ pos,
    const float* __restrict__ nb1, const float* __restrict__ nb2,
    const __bf16* __restrict__ nw1p, const __bf16* __restrict__ nw2p,
    const float* __restrict__ lng, const float* __restrict__ lnb,
    const float* __restrict__ aggMsg, const float* __restrict__ aggPos,
    float* __restrict__ hOut, float* __restrict__ posOut,
    int N, int nTiles) {
  __shared__ __bf16 Ast_s[4][16 * 256];   // [h | agg_msg] stage; reused as fp32 X
  __shared__ __bf16 Ms_s[4][16 * 128];

  const int lane = threadIdx.x & 31;
  const int wid  = threadIdx.x >> 5;
  const int tile = blockIdx.x * 4 + wid;
  if (tile >= nTiles) return;

  __bf16* Ast = &Ast_s[wid][0];
  __bf16* Ms  = &Ms_s[wid][0];
  const int hi = lane >> 4;
  const int lo = lane & 15;
  const int n0 = tile * 16;

  // Stage [h[i] | aggMsg[i]] -> bf16 [16][256]
#pragma unroll 4
  for (int it = 0; it < 32; ++it) {
    int c4   = it * 32 + lane;
    int nIdx = c4 >> 6;
    int kc   = c4 & 63;
    int node = n0 + nIdx;
    if (node >= N) node = N - 1;
    const float4* src = (kc < 32)
        ? (const float4*)(h + (size_t)node * H + kc * 4)
        : (const float4*)(aggMsg + (size_t)node * H + (kc - 32) * 4);
    float4 v = *src;
    v4bf o = { (__bf16)v.x, (__bf16)v.y, (__bf16)v.z, (__bf16)v.w };
    *(v4bf*)(Ast + nIdx * 256 + kc * 4) = o;
  }
  lds_fence();

  // node layer 1
#pragma unroll
  for (int nt = 0; nt < 8; ++nt) {
    int Ncol = nt * 16 + lo;
    float b1 = nb1[Ncol];
    v8f c;
#pragma unroll
    for (int v = 0; v < 8; ++v) c[v] = b1;
    c = wmma_tiles(Ast, 256, nw1p, nt, 8, c, lane);
#pragma unroll
    for (int v = 0; v < 8; ++v)
      Ms[(v + 8 * hi) * H + Ncol] = (__bf16)silu_f(c[v]);
  }
  lds_fence();

  // node layer 2 + residual -> fp32 X stage (reuses Ast region)
  float* Xs = (float*)Ast;
#pragma unroll
  for (int nt = 0; nt < 8; ++nt) {
    int Ncol = nt * 16 + lo;
    float b2 = nb2[Ncol];
    v8f c;
#pragma unroll
    for (int v = 0; v < 8; ++v) c[v] = b2;
    c = wmma_tiles(Ms, 128, nw2p, nt, 4, c, lane);
#pragma unroll
    for (int v = 0; v < 8; ++v) {
      int Mr = v + 8 * hi;
      int node = n0 + Mr;
      float hv = (node < N) ? h[(size_t)node * H + Ncol] : 0.f;
      Xs[Mr * H + Ncol] = c[v] + hv;
    }
  }
  lds_fence();

  // layernorm + outputs: one row per lane (lanes 0..15)
  if (lane < 16) {
    int node = n0 + lane;
    if (node < N) {
      const float4* xr = (const float4*)(Xs + lane * H);
      float sum = 0.f, sq = 0.f;
#pragma unroll 8
      for (int i = 0; i < 32; ++i) {
        float4 x = xr[i];
        sum += x.x + x.y + x.z + x.w;
        sq  += x.x * x.x + x.y * x.y + x.z * x.z + x.w * x.w;
      }
      float mu  = sum * (1.f / 128.f);
      float var = sq * (1.f / 128.f) - mu * mu;
      float rs  = rsqrtf(var + LN_EPS);
      const float4* gr = (const float4*)lng;
      const float4* br = (const float4*)lnb;
      float4* outr = (float4*)(hOut + (size_t)node * H);
#pragma unroll 8
      for (int i = 0; i < 32; ++i) {
        float4 x = xr[i], g = gr[i], b = br[i];
        float4 o;
        o.x = (x.x - mu) * rs * g.x + b.x;
        o.y = (x.y - mu) * rs * g.y + b.y;
        o.z = (x.z - mu) * rs * g.z + b.z;
        o.w = (x.w - mu) * rs * g.w + b.w;
        outr[i] = o;
      }
      posOut[(size_t)node * 3 + 0] = pos[(size_t)node * 3 + 0] + aggPos[(size_t)node * 3 + 0];
      posOut[(size_t)node * 3 + 1] = pos[(size_t)node * 3 + 1] + aggPos[(size_t)node * 3 + 1];
      posOut[(size_t)node * 3 + 2] = pos[(size_t)node * 3 + 2] + aggPos[(size_t)node * 3 + 2];
    }
  }
}

extern "C" void kernel_launch(void* const* d_in, const int* in_sizes, int n_in,
                              void* d_out, int out_size, void* d_ws, size_t ws_size,
                              hipStream_t stream) {
  (void)n_in; (void)out_size; (void)ws_size;
  const float*     h   = (const float*)d_in[0];
  const float*     pos = (const float*)d_in[1];
  const long long* ei  = (const long long*)d_in[2];
  const float* ew1 = (const float*)d_in[3];
  const float* eb1 = (const float*)d_in[4];
  const float* ew2 = (const float*)d_in[5];
  const float* eb2 = (const float*)d_in[6];
  const float* nw1 = (const float*)d_in[7];
  const float* nb1 = (const float*)d_in[8];
  const float* nw2 = (const float*)d_in[9];
  const float* nb2 = (const float*)d_in[10];
  const float* cw1 = (const float*)d_in[11];
  const float* cb1 = (const float*)d_in[12];
  const float* cw2 = (const float*)d_in[13];
  const float* lng = (const float*)d_in[14];
  const float* lnb = (const float*)d_in[15];

  const int N = in_sizes[0] / H;
  const int E = in_sizes[2] / 2;

  char* ws = (char*)d_ws;
  size_t offPos = (size_t)N * H * sizeof(float);
  size_t offW   = (offPos + (size_t)N * 3 * sizeof(float) + 255) & ~(size_t)255;
  float*  aggMsg = (float*)(ws + 0);
  float*  aggPos = (float*)(ws + offPos);
  __bf16* ew1p = (__bf16*)(ws + offW);
  __bf16* ew2p = ew1p + 256 * H;
  __bf16* cw1p = ew2p + 128 * H;
  __bf16* nw1p = cw1p + 128 * H;
  __bf16* nw2p = nw1p + 256 * H;

  // zero fp32 accumulators every launch (graph-capture safe)
  hipMemsetAsync(d_ws, 0, offPos + (size_t)N * 3 * sizeof(float), stream);

  egnn_pack_w<<<(256 * H + 255) / 256, 256, 0, stream>>>(ew1, ew1p, 256);
  egnn_pack_w<<<(128 * H + 255) / 256, 256, 0, stream>>>(ew2, ew2p, 128);
  egnn_pack_w<<<(128 * H + 255) / 256, 256, 0, stream>>>(cw1, cw1p, 128);
  egnn_pack_w<<<(256 * H + 255) / 256, 256, 0, stream>>>(nw1, nw1p, 256);
  egnn_pack_w<<<(128 * H + 255) / 256, 256, 0, stream>>>(nw2, nw2p, 128);

  int eTiles = (E + 15) / 16;
  egnn_edge_kernel<<<(eTiles + 3) / 4, 128, 0, stream>>>(
      h, pos, ei, ew1, eb1, eb2, cb1, cw2, ew1p, ew2p, cw1p, aggMsg, aggPos,
      E, eTiles);

  int nTiles = (N + 15) / 16;
  float* hOut   = (float*)d_out;
  float* posOut = hOut + (size_t)N * H;
  egnn_node_kernel<<<(nTiles + 3) / 4, 128, 0, stream>>>(
      h, pos, nb1, nb2, nw1p, nw2p, lng, lnb, aggMsg, aggPos, hOut, posOut,
      N, nTiles);
}
